// Camera2MapMulti_78993038508423
// MI455X (gfx1250) — matrix operand order repaired and checked by reading, not compiled, not run on hardware
//
#include <hip/hip_runtime.h>
#include <hip/hip_bf16.h>

typedef __attribute__((ext_vector_type(16))) _Float16 v16h;
typedef __attribute__((ext_vector_type(8)))  float    v8f;

namespace {
constexpr int Hc = 160, Wc = 320, HWc = Hc * Wc;   // per-image pixels
constexpr int BNc = 4;                              // B*N
constexpr int Fc = 32, ZDc = 16, CIN = 48, COUT = 32;
constexpr int KPAD = 448;                           // 432 padded to 14*32
constexpr int KSTEPS = KPAD / 32;                   // 14
constexpr int GHc = 256, GWc = 256, NVOX = GHc * GWc;
constexpr int BAc = 2;                              // B * (N/NC)
constexpr int NPTS = 2 * HWc;                       // points per ba

// halo-padded pixel-major staging: [bn][H+2][W+2][48ch], 96 B per pixel
constexpr int PH = Hc + 2, PW = Wc + 2, PLANE = PH * PW;  // 162*322 = 52164

// ---- workspace layout (bytes) ----
constexpr size_t OFF_VOLF  = 0;                                       // float[2*65536*32]
constexpr size_t OFF_DENS  = OFF_VOLF + (size_t)BAc * NVOX * COUT * 4;// float[2*65536]
constexpr size_t OFF_XPIX  = OFF_DENS + (size_t)BAc * NVOX * 4;       // half[4*PLANE*48]
constexpr size_t OFF_ZPAD  = OFF_XPIX + (size_t)BNc * PLANE * CIN * 2;// half[16] zero pad
constexpr size_t ZERO_END  = OFF_ZPAD + 32;                           // zeroed range end
constexpr size_t OFF_MASK  = ZERO_END;                                // float[4*HW]
constexpr size_t OFF_APACK = OFF_MASK + (size_t)BNc * HWc * 4;        // half[2*14*32*16]
} // namespace

// ------------------------------------------------------------------
// zeroes volf + dens accumulators AND the halo-padded staging + zero pad
__global__ void zero_ws_kernel(float* p, int nf) {
  int i = blockIdx.x * blockDim.x + threadIdx.x;
  if (i < nf) p[i] = 0.0f;
}

// Pack conv_w [32,48,3,3] into WMMA A fragments with K reordered tap-major:
// kidx = r*48 + ci  (r = ky*3+kx tap, ci = input channel), padded to 448.
// A 16x32 K order used here (B matches): K_in_step = h + 16*(lane/16), M = lane%16.
__global__ void pack_weights_kernel(const float* __restrict__ conv_w,
                                    _Float16* __restrict__ Apack) {
  int i = blockIdx.x * blockDim.x + threadIdx.x;
  if (i >= 2 * KSTEPS * 32) return;
  int lane = i & 31;
  int rem  = i >> 5;
  int kk   = rem % KSTEPS;
  int t    = rem / KSTEPS;
  int g    = lane >> 4;
  int co   = t * 16 + (lane & 15);
  _Float16* dst = Apack + (size_t)i * 16;
#pragma unroll
  for (int h = 0; h < 16; ++h) {
    int kin  = h + g * 16;
    int kidx = kk * 32 + kin;         // tap-major K index
    int r    = kidx / 48;             // 0..9 (9 => padding)
    int ci   = kidx - r * 48;
    float v = (r < 9) ? conv_w[co * 432 + ci * 9 + r] : 0.0f;
    dst[h] = (_Float16)v;
  }
}

// Wave-structured per-pixel prep: 16 pixels per wave.
// - backproject + z-MLP layer1 (each lane computes the 16 h1 values of its
//   A-fragment slots), layer2 done with ONE v_wmma_f32_16x16x32_f16
// - stage fp16 feats+zf pixel-major with halo
// - mask / xy output / density bilinear splat on the g==0 half-wave
__global__ void prep_pixels_kernel(const float* __restrict__ depth,
                                   const float* __restrict__ feats,
                                   const float* __restrict__ p2p,
                                   const float* __restrict__ z_w1,
                                   const float* __restrict__ z_b1,
                                   const float* __restrict__ z_w2,
                                   const float* __restrict__ z_b2,
                                   _Float16* __restrict__ Xpix,
                                   float* __restrict__ maskF,
                                   float* __restrict__ xyOut,
                                   float* __restrict__ densAcc) {
  int wave = (blockIdx.x * blockDim.x + threadIdx.x) >> 5;
  int lane = threadIdx.x & 31;
  int n = lane & 15, g = lane >> 4;
  int segPerImg = HWc / 16;
  int bn  = wave / segPerImg;
  int p0  = (wave - bn * segPerImg) * 16;
  int hrow = p0 / Wc;
  int w0   = p0 - hrow * Wc;        // W multiple of 16: segment stays in one row
  int wcol = w0 + n;
  int idx  = bn * HWc + p0 + n;

  float d = depth[idx];
  const float* M = p2p + bn * 16;
  float u = (float)wcol * d, v = (float)hrow * d;
  float x0 = M[0]*u + M[1]*v + M[2]*d  + M[3];
  float x1 = M[4]*u + M[5]*v + M[6]*d  + M[7];
  float x2 = M[8]*u + M[9]*v + M[10]*d + M[11];

  // layer-1 (1->32, relu): fill this lane's A-fragment slots
  v16h a;
#pragma unroll
  for (int e = 0; e < 16; ++e) {
    int j = e + g * 16;   // same kin mapping as A layout
    a[e] = (_Float16)fmaxf(x2 * z_w1[j] + z_b1[j], 0.0f);
  }
  // layer-2 B fragment: B[k][nc] = z_w2[nc*32+k]; lane element e -> K=e+16g, N=n
  v16h bf;
#pragma unroll
  for (int e = 0; e < 16; ++e)
    bf[e] = (_Float16)z_w2[n * 32 + g * 16 + e];

  v8f zacc = {};
  zacc = __builtin_amdgcn_wmma_f32_16x16x32_f16(false, a, false, bf, (short)0, zacc, false, false);

  // write zf: lane holds output channel n for pixels m = q + 8g
  float zb = z_b2[n];
  size_t rowBase = (size_t)bn * PLANE + (size_t)(hrow + 1) * PW;  // padded row
#pragma unroll
  for (int q = 0; q < 8; ++q) {
    int m = q + g * 8;
    float zv = fmaxf(zacc[q] + zb, 0.0f);
    Xpix[(rowBase + (w0 + m + 1)) * CIN + Fc + n] = (_Float16)zv;
  }

  // stage feats: lane stages channels g*16..g*16+15 of pixel n (contiguous 32B)
  {
    const float* fb = feats + (size_t)bn * Fc * HWc + p0 + n;
    _Float16* Xb = Xpix + (rowBase + (wcol + 1)) * CIN + g * 16;
#pragma unroll
    for (int ci = 0; ci < 16; ++ci)
      Xb[ci] = (_Float16)fb[(size_t)(g * 16 + ci) * HWc];
  }

  // per-pixel outputs on half-wave only (after the WMMA; EXEC was all-ones)
  if (g == 0) {
    bool m = (x0 >= -25.6f) && (x0 < 25.6f) &&
             (x1 >= -25.6f) && (x1 < 25.6f) &&
             (x2 >= -3.0f)  && (x2 < 1.0f);
    maskF[idx] = m ? 1.0f : 0.0f;

    float xg = (25.6f - x1) * 5.0f;
    float yg = (25.6f - x0) * 5.0f;
    int ba = bn >> 1;
    int pt = (bn & 1) * HWc + p0 + n;
    float* xo = xyOut + ((size_t)ba * NPTS + pt) * 2;
    xo[0] = xg;
    xo[1] = yg;

    float fx = floorf(xg), fy = floorf(yg);
    int Xi = (int)fx, Yi = (int)fy;
    float rX = xg - fx, rY = yg - fy;
    float* dacc = densAcc + (size_t)ba * NVOX;
#pragma unroll
    for (int dx = 0; dx < 2; ++dx) {
#pragma unroll
      for (int dy = 0; dy < 2; ++dy) {
        int Xc = Xi + dx, Yc = Yi + dy;
        if (Xc >= 0 && Xc < GWc && Yc >= 0 && Yc < GHc) {
          float wX = dx ? rX : (1.0f - rX);
          float wY = dy ? rY : (1.0f - rY);
          atomicAdd(dacc + Yc * GWc + Xc, wX * wY);
        }
      }
    }
  }
}

// Implicit-GEMM 3x3 conv via v_wmma_f32_16x16x32_f16 + fused bias/relu/mask
// + bilinear feature splat. One wave = 16 pixels x 32 channels.
// Tap-major K + pixel-major halo staging => each B fragment is ONE aligned
// 32-byte v16h load (no gather, no boundary checks).
__global__ void conv_wmma_splat_kernel(const _Float16* __restrict__ Xpix,
                                       const _Float16* __restrict__ Apack,
                                       const _Float16* __restrict__ zeroPad,
                                       const float* __restrict__ maskF,
                                       const float* __restrict__ conv_b,
                                       const float* __restrict__ xyOut,
                                       float* __restrict__ volfAcc) {
  int wave = (blockIdx.x * blockDim.x + threadIdx.x) >> 5;
  int lane = threadIdx.x & 31;
  int segPerImg = HWc / 16;                 // 3200
  int bn  = wave / segPerImg;
  int seg = wave - bn * segPerImg;
  int p0  = seg * 16;
  int h0  = p0 / Wc;
  int w0  = p0 - h0 * Wc;                   // row-aligned segment
  int n   = lane & 15;                      // pixel within segment
  int g   = lane >> 4;                      // lane group
  int wp  = w0 + n;

  const _Float16* Xbn = Xpix + (size_t)bn * PLANE * CIN;  // wave-uniform base

  v8f acc0 = {};
  v8f acc1 = {};
#pragma unroll
  for (int kk = 0; kk < KSTEPS; ++kk) {
    v16h a0 = *(const v16h*)(Apack + ((size_t)((0 * KSTEPS + kk) * 32 + lane)) * 16);
    v16h a1 = *(const v16h*)(Apack + ((size_t)((1 * KSTEPS + kk) * 32 + lane)) * 16);

    // q enumerates 16-wide K blocks: q = 2*kk + g in 0..27 (27 = zero padding)
    int q   = kk * 2 + g;
    int r   = q / 3;                        // filter tap 0..8 (9 = padding)
    int ci0 = (q - r * 3) << 4;             // 0,16,32
    int dy  = r / 3;
    int dx  = r - dy * 3;
    const _Float16* bp = (q < 27)
        ? Xbn + ((size_t)(h0 + dy) * PW + (wp + dx)) * CIN + ci0
        : zeroPad;
    v16h b = *(const v16h*)bp;

    acc0 = __builtin_amdgcn_wmma_f32_16x16x32_f16(false, a0, false, b, (short)0, acc0, false, false);
    acc1 = __builtin_amdgcn_wmma_f32_16x16x32_f16(false, a1, false, b, (short)0, acc1, false, false);
  }

  // epilogue: bias + relu + mask, then bilinear feature splat
  float mask = maskF[bn * HWc + p0 + n];
  int ba = bn >> 1;
  int pt = (bn & 1) * HWc + p0 + n;
  const float* xyp = xyOut + ((size_t)ba * NPTS + pt) * 2;
  float xg = xyp[0], yg = xyp[1];
  float fx = floorf(xg), fy = floorf(yg);
  int Xi = (int)fx, Yi = (int)fy;
  float rX = xg - fx, rY = yg - fy;
  float* acc = volfAcc + (size_t)ba * NVOX * COUT;

#pragma unroll
  for (int t = 0; t < 2; ++t) {
    v8f cc = t ? acc1 : acc0;
#pragma unroll
    for (int q = 0; q < 8; ++q) {
      // C/D layout: VGPR q, lanes 0-15 -> M=q, lanes 16-31 -> M=q+8
      int co = t * 16 + g * 8 + q;
      float y = fmaxf(cc[q] + conv_b[co], 0.0f) * mask;
#pragma unroll
      for (int dx2 = 0; dx2 < 2; ++dx2) {
        int Xc = Xi + dx2;
        float wX = dx2 ? rX : (1.0f - rX);
#pragma unroll
        for (int dy2 = 0; dy2 < 2; ++dy2) {
          int Yc = Yi + dy2;
          float wY = dy2 ? rY : (1.0f - rY);
          if (Xc >= 0 && Xc < GWc && Yc >= 0 && Yc < GHc) {
            atomicAdd(acc + ((size_t)(Yc * GWc + Xc)) * COUT + co, wX * wY * y);
          }
        }
      }
    }
  }
}

// scatter-mean + output layout
__global__ void finalize_kernel(const float* __restrict__ densAcc,
                                const float* __restrict__ volfAcc,
                                float* __restrict__ outF,
                                float* __restrict__ outD) {
  int i = blockIdx.x * blockDim.x + threadIdx.x;
  if (i >= BAc * NVOX) return;
  int ba  = i >> 16;
  int vox = i & (NVOX - 1);
  float dens = densAcc[i];
  outD[i] = dens;
  float inv = 1.0f / fmaxf(dens, 1.0f);
  const float* src = volfAcc + (size_t)i * COUT;
  float* dst = outF + (size_t)ba * COUT * NVOX + vox;
#pragma unroll
  for (int co = 0; co < COUT; ++co) dst[(size_t)co * NVOX] = src[co] * inv;
}

// ------------------------------------------------------------------
extern "C" void kernel_launch(void* const* d_in, const int* in_sizes, int n_in,
                              void* d_out, int out_size, void* d_ws, size_t ws_size,
                              hipStream_t stream) {
  const float* depth  = (const float*)d_in[0];
  const float* feats  = (const float*)d_in[1];
  const float* p2p    = (const float*)d_in[2];
  const float* z_w1   = (const float*)d_in[3];
  const float* z_b1   = (const float*)d_in[4];
  const float* z_w2   = (const float*)d_in[5];
  const float* z_b2   = (const float*)d_in[6];
  const float* conv_w = (const float*)d_in[7];
  const float* conv_b = (const float*)d_in[8];

  char* ws = (char*)d_ws;
  float*    volfAcc = (float*)(ws + OFF_VOLF);
  float*    densAcc = (float*)(ws + OFF_DENS);
  _Float16* Xpix    = (_Float16*)(ws + OFF_XPIX);
  _Float16* zeroPad = (_Float16*)(ws + OFF_ZPAD);
  float*    maskF   = (float*)(ws + OFF_MASK);
  _Float16* Apack   = (_Float16*)(ws + OFF_APACK);

  float* outF  = (float*)d_out;                       // [2,32,256,256]
  float* outD  = outF + (size_t)BAc * COUT * NVOX;    // [2,1,256,256]
  float* outXY = outD + (size_t)BAc * NVOX;           // [2,102400,2]

  // zero accumulators + halo-padded staging + zero pad
  int nz = (int)(ZERO_END / 4);
  zero_ws_kernel<<<(nz + 255) / 256, 256, 0, stream>>>(volfAcc, nz);

  pack_weights_kernel<<<4, 256, 0, stream>>>(conv_w, Apack);

  int nwaves = BNc * (HWc / 16);                // 12800 waves (16 px each)
  prep_pixels_kernel<<<nwaves / 8, 256, 0, stream>>>(
      depth, feats, p2p, z_w1, z_b1, z_w2, z_b2, Xpix, maskF, outXY, densAcc);

  conv_wmma_splat_kernel<<<nwaves / 8, 256, 0, stream>>>(
      Xpix, Apack, zeroPad, maskF, conv_b, outXY, volfAcc);

  int nvoxTot = BAc * NVOX;
  finalize_kernel<<<(nvoxTot + 255) / 256, 256, 0, stream>>>(
      densAcc, volfAcc, outF, outD);
}
